// SuccessiveSwinTransfomerBlock_11184094839189
// MI455X (gfx1250) — matrix operand order repaired and checked by reading, not compiled, not run on hardware
//
#include <hip/hip_runtime.h>

// ---------------------------------------------------------------------------
// CDNA5 (gfx1250) Swin block pipeline
//   - bf16 WMMA (v_wmma_f32_16x16x32_bf16) for all GEMMs / attention matmuls
//   - Tensor Data Mover (tensor_load_to_lds) for GEMM A-tile staging, with
//     hardware LDS padding reproducing the 136-bf16 bank-conflict stride
// ---------------------------------------------------------------------------

typedef __attribute__((ext_vector_type(16))) __bf16 bf16x16;
typedef __attribute__((ext_vector_type(8)))  float  f32x8;
typedef unsigned int u32x4 __attribute__((ext_vector_type(4)));
typedef int          i32x4 __attribute__((ext_vector_type(4)));
typedef int          i32x8 __attribute__((ext_vector_type(8)));

#define M_TOT 200704   // B * S = 64 * 3136
#define K_DIM 128      // embedding dim

#if __has_builtin(__builtin_amdgcn_tensor_load_to_lds) && \
    __has_builtin(__builtin_amdgcn_s_wait_tensorcnt)
#define HAVE_TDM 1
#else
#define HAVE_TDM 0
#endif

union FragB { bf16x16 v; uint4 q[2]; unsigned short u[16]; };
union AccF  { f32x8 v; float f[8]; };

__device__ __forceinline__ unsigned short f2bf(float f) {
    union { float f; unsigned u; } v; v.f = f;
    unsigned r = v.u + 0x7FFFu + ((v.u >> 16) & 1u);   // round-to-nearest-even
    return (unsigned short)(r >> 16);
}

// A-matrix fragment (16 x 32 bf16). ISA 7.12.2: lane m (0-15) holds
// K {h*8..h*8+7, 16+h*8..16+h*8+7}, h = lane>>4. base points at (m0, k0).
__device__ __forceinline__ bf16x16 load_frag_a(const unsigned short* base,
                                               int stride, int lane) {
    int m = lane & 15, h = lane >> 4;
    FragB f;
    f.q[0] = *(const uint4*)(base + m * stride + h * 8);
    f.q[1] = *(const uint4*)(base + m * stride + 16 + h * 8);
    return f.v;
}

// B-matrix fragment (32 x 16 bf16) from a source stored rows[n][k]
// (contraction contiguous): lane n holds K {h*8..h*8+7, 16+h*8..16+h*8+7}.
__device__ __forceinline__ bf16x16 load_frag_b(const unsigned short* base,
                                               int stride, int lane) {
    int n = lane & 15, h = lane >> 4;
    FragB f;
    f.q[0] = *(const uint4*)(base + n * stride + h * 8);
    f.q[1] = *(const uint4*)(base + n * stride + 16 + h * 8);
    return f.v;
}

__device__ __forceinline__ f32x8 wmma_bf16(bf16x16 a, bf16x16 b, f32x8 c) {
    return __builtin_amdgcn_wmma_f32_16x16x32_bf16(
        false, a, false, b, (short)0, c, false, false);
}

// ---------------------------------------------------------------------------
// LayerNorm (D=128) -> bf16. One wave per row; 4 f32 per lane; wave32 shuffles.
// ---------------------------------------------------------------------------
__global__ __launch_bounds__(256) void ln_bf16_kernel(
    const float* __restrict__ x, const float* __restrict__ gamma,
    const float* __restrict__ beta, unsigned short* __restrict__ y) {
    int wv = threadIdx.x >> 5, lane = threadIdx.x & 31;
    size_t row = (size_t)blockIdx.x * 8 + wv;
    const float* xr = x + row * K_DIM;
    float4 v = *(const float4*)(xr + lane * 4);
    float s = v.x + v.y + v.z + v.w;
#pragma unroll
    for (int o = 16; o > 0; o >>= 1) s += __shfl_xor(s, o, 32);
    float mu = s * 0.0078125f;
    float d0 = v.x - mu, d1 = v.y - mu, d2 = v.z - mu, d3 = v.w - mu;
    float q = d0 * d0 + d1 * d1 + d2 * d2 + d3 * d3;
#pragma unroll
    for (int o = 16; o > 0; o >>= 1) q += __shfl_xor(q, o, 32);
    float rs = rsqrtf(q * 0.0078125f + 1e-5f);
    float4 g = *(const float4*)(gamma + lane * 4);
    float4 b = *(const float4*)(beta + lane * 4);
    unsigned short* yr = y + row * K_DIM + lane * 4;
    yr[0] = f2bf(d0 * rs * g.x + b.x);
    yr[1] = f2bf(d1 * rs * g.y + b.y);
    yr[2] = f2bf(d2 * rs * g.z + b.z);
    yr[3] = f2bf(d3 * rs * g.w + b.w);
}

// ---------------------------------------------------------------------------
// GEMM: out[M,N] = A_bf16[M,128] @ W_f32[N,128]^T + bias (+ residual_f32)
// 128x64 tile per WG (256 thr = 8 waves, 4x2 wave grid, 32x32 per wave).
// A-tile staged via TDM (one DMA per workgroup, HW-padded LDS rows).
// ---------------------------------------------------------------------------
template <bool OUT_BF16, bool HAS_RES>
__global__ __launch_bounds__(256) void gemm_bf16_kernel(
    const unsigned short* __restrict__ A, const float* __restrict__ W,
    const float* __restrict__ bias, const float* __restrict__ res,
    void* __restrict__ out, int N) {
    __shared__ unsigned short As[128 * 136];
    __shared__ unsigned short Bs[64 * 136];
    int tid = threadIdx.x;
    int m0 = blockIdx.x * 128;
    int n0 = blockIdx.y * 64;

#if HAVE_TDM
    // --- A tile via Tensor Data Mover -------------------------------------
    // D# (ISA 8.3/8.4): 2D tile 128x128, data_size=1 (2B elems),
    // pad_interval=64 dwords (one 256B row), pad_amount=4 dwords (16B)
    // -> LDS row stride 272B = 136 bf16, matching the fragment loads.
    if (tid == 0) {
        unsigned lds_off = (unsigned)(size_t)(void*)As;
        unsigned long long ga =
            (unsigned long long)(const void*)(A + (size_t)m0 * K_DIM);
        u32x4 g0 = {0u, 0u, 0u, 0u};
        g0[0] = 1u;                                        // count=1, user mode
        g0[1] = lds_off;                                   // lds_addr
        g0[2] = (unsigned)(ga & 0xFFFFFFFFull);            // global_addr[31:0]
        g0[3] = (unsigned)((ga >> 32) & 0x1FFFFFFull) | (2u << 30); // +type=2
        unsigned td0 = 128u, td1 = (unsigned)M_TOT;        // tensor dims
        unsigned tl0 = 128u, tl1 = 128u, tl2 = 0u;         // tile dims
        unsigned long long st0 = 128ull, st1 = 0ull;       // strides (elems)
        i32x8 g1 = {0, 0, 0, 0, 0, 0, 0, 0};
        g1[0] = (int)((1u << 16)            // data_size = 2 bytes
                    | (1u << 20)            // pad_enable
                    | (5u << 22)            // pad_interval: 64 dwords
                    | (3u << 25));          // pad_amount:   4 dwords
        g1[1] = (int)((td0 & 0xFFFFu) << 16);
        g1[2] = (int)(((td0 >> 16) & 0xFFFFu) | ((td1 & 0xFFFFu) << 16));
        g1[3] = (int)(((td1 >> 16) & 0xFFFFu) | (tl0 << 16));
        g1[4] = (int)(tl1 | (tl2 << 16));
        g1[5] = (int)(st0 & 0xFFFFFFFFull);
        g1[6] = (int)(((st0 >> 32) & 0xFFFFull) |
                      ((unsigned)(st1 & 0xFFFFull) << 16));
        g1[7] = (int)((st1 >> 16) & 0xFFFFFFFFull);
        i32x4 z4 = {0, 0, 0, 0};
#if defined(__clang_major__) && __clang_major__ >= 23
        i32x8 z8 = {0, 0, 0, 0, 0, 0, 0, 0};
        __builtin_amdgcn_tensor_load_to_lds(g0, g1, z4, z4, z8, 0);
#else
        __builtin_amdgcn_tensor_load_to_lds(g0, g1, z4, z4, 0);
#endif
    }
#else
    // --- fallback: cooperative A-tile copy --------------------------------
    for (int i = tid; i < 128 * 16; i += 256) {
        int r = i >> 4, c = (i & 15) * 8;
        *(uint4*)&As[r * 136 + c] =
            *(const uint4*)&A[(size_t)(m0 + r) * K_DIM + c];
    }
#endif

    // B tile: 64 rows x 128, convert f32 -> bf16 (weights stay resident in L2)
    for (int i = tid; i < 64 * 32; i += 256) {
        int r = i >> 5, c = (i & 31) * 4;
        float4 w = *(const float4*)&W[(size_t)(n0 + r) * K_DIM + c];
        unsigned short* d = &Bs[r * 136 + c];
        d[0] = f2bf(w.x); d[1] = f2bf(w.y); d[2] = f2bf(w.z); d[3] = f2bf(w.w);
    }
#if HAVE_TDM
    __builtin_amdgcn_s_wait_tensorcnt(0);  // per-wave counter; no-op elsewhere
#endif
    __syncthreads();

    int wv = tid >> 5, lane = tid & 31;
    int wrow = wv >> 1, wcol = wv & 1;
    f32x8 acc[2][2] = {};
#pragma unroll
    for (int k = 0; k < K_DIM; k += 32) {
        bf16x16 a0 = load_frag_a(&As[(wrow * 32 + 0) * 136 + k], 136, lane);
        bf16x16 a1 = load_frag_a(&As[(wrow * 32 + 16) * 136 + k], 136, lane);
        bf16x16 b0 = load_frag_b(&Bs[(wcol * 32 + 0) * 136 + k], 136, lane);
        bf16x16 b1 = load_frag_b(&Bs[(wcol * 32 + 16) * 136 + k], 136, lane);
        acc[0][0] = wmma_bf16(a0, b0, acc[0][0]);
        acc[0][1] = wmma_bf16(a0, b1, acc[0][1]);
        acc[1][0] = wmma_bf16(a1, b0, acc[1][0]);
        acc[1][1] = wmma_bf16(a1, b1, acc[1][1]);
    }

    int ln = lane & 15, lh = lane >> 4;
#pragma unroll
    for (int mt = 0; mt < 2; ++mt)
#pragma unroll
        for (int nt = 0; nt < 2; ++nt) {
            AccF c; c.v = acc[mt][nt];
            int col = n0 + wcol * 32 + nt * 16 + ln;
            float bv = bias[col];
#pragma unroll
            for (int r = 0; r < 8; ++r) {
                int row = m0 + wrow * 32 + mt * 16 + lh * 8 + r;
                float v = c.f[r] + bv;
                if (HAS_RES) v += res[(size_t)row * N + col];
                if (OUT_BF16)
                    ((unsigned short*)out)[(size_t)row * N + col] = f2bf(v);
                else
                    ((float*)out)[(size_t)row * N + col] = v;
            }
        }
}

// ---------------------------------------------------------------------------
// Windowed attention: one WG (128 thr = 4 waves) per (window, head).
// Roll / window permutation folded into the gather/scatter indexing.
// ---------------------------------------------------------------------------
__global__ __launch_bounds__(128) void win_attn_kernel(
    const unsigned short* __restrict__ qkv,   // [M][384] bf16
    const float* __restrict__ table,          // [169][4]
    unsigned short* __restrict__ attn,        // [M][128] bf16
    int shifted) {
    __shared__ unsigned short Qs[64 * 32];
    __shared__ unsigned short Ks[64 * 32];
    __shared__ unsigned short Vt[32 * 64];    // V transposed: [ch][token]
    __shared__ float Sc[64 * 64];
    __shared__ unsigned short Ps[64 * 64];

    int blk = blockIdx.x;            // = (b*64 + w)*4 + h (matches ref batching)
    int h = blk & 3;
    int bw = blk >> 2;
    int w = bw & 63, b = bw >> 6;
    int a = w >> 3, c = w & 7;
    int tid = threadIdx.x;

    for (int i = tid; i < 64 * 32; i += 128) { Qs[i] = 0; Ks[i] = 0; Vt[i] = 0; }
    __syncthreads();

    // Gather Q/K rows (49 tokens x 32 ch, 8-bf16 chunks)
    for (int i = tid; i < 49 * 4; i += 128) {
        int t = i >> 2, ch = (i & 3) * 8;
        int bq = t / 7, d = t - bq * 7;
        int rr = a * 7 + bq, cc = c * 7 + d;
        if (shifted) { rr = (rr + 3) % 56; cc = (cc + 3) % 56; }
        size_t row = (size_t)b * 3136 + (size_t)rr * 56 + cc;
        const unsigned short* base = qkv + row * 384 + h * 32;
        *(uint4*)&Qs[t * 32 + ch] = *(const uint4*)(base + ch);
        *(uint4*)&Ks[t * 32 + ch] = *(const uint4*)(base + 128 + ch);
    }
    // Gather V transposed (B-fragment wants contraction contiguous)
    for (int i = tid; i < 49 * 32; i += 128) {
        int t = i >> 5, ch = i & 31;
        int bq = t / 7, d = t - bq * 7;
        int rr = a * 7 + bq, cc = c * 7 + d;
        if (shifted) { rr = (rr + 3) % 56; cc = (cc + 3) % 56; }
        size_t row = (size_t)b * 3136 + (size_t)rr * 56 + cc;
        Vt[ch * 64 + t] = qkv[row * 384 + 256 + h * 32 + ch];
    }
    __syncthreads();

    int wv = tid >> 5, lane = tid & 31;
    int ln = lane & 15, lh = lane >> 4;

    // scores: wave wv covers query rows [wv*16, wv*16+16), all 64 key cols
    {
        bf16x16 aq = load_frag_a(&Qs[(wv * 16) * 32], 32, lane);  // hd = 32
#pragma unroll
        for (int nt = 0; nt < 4; ++nt) {
            bf16x16 bk = load_frag_b(&Ks[(nt * 16) * 32], 32, lane);
            f32x8 acc = {};
            acc = wmma_bf16(aq, bk, acc);
            AccF cf; cf.v = acc;
#pragma unroll
            for (int r = 0; r < 8; ++r)
                Sc[(wv * 16 + lh * 8 + r) * 64 + nt * 16 + ln] = cf.f[r];
        }
    }
    __syncthreads();

    // softmax, thread-per-row; scale/bias/mask applied here
    if (tid < 64) {
        int qt = tid;
        if (qt >= 49) {
            for (int k = 0; k < 64; ++k) Ps[qt * 64 + k] = 0;
        } else {
            const float scale = 0.17677669529663687f;  // 1/sqrt(32)
            int i1 = qt / 7, j1 = qt - i1 * 7;
            int wm = ((w << 2) + h) & 63;   // faithful tile(mask,(NH,1,1)) index
            int am = wm >> 3, cm = wm & 7;
            int regq = 0;
            if (shifted) {
                int ur = (am * 7 + i1 + 3) % 56, uc = (cm * 7 + j1 + 3) % 56;
                regq = ((ur + 3) / 7) * 8 + ((uc + 3) / 7);
            }
            float mx = -1e30f;
            for (int k = 0; k < 49; ++k) {
                int i2 = k / 7, j2 = k - i2 * 7;
                int ri = (i1 - i2 + 6) * 13 + (j1 - j2 + 6);
                float v = Sc[qt * 64 + k] * scale + table[ri * 4 + h];
                if (shifted) {
                    int ur = (am * 7 + i2 + 3) % 56, uc = (cm * 7 + j2 + 3) % 56;
                    int regk = ((ur + 3) / 7) * 8 + ((uc + 3) / 7);
                    if (regk != regq) v += 1e-9f;   // faithful (1-eq)*1e-9
                }
                Sc[qt * 64 + k] = v;
                mx = fmaxf(mx, v);
            }
            float sum = 0.f;
            for (int k = 0; k < 49; ++k) {
                float e = __expf(Sc[qt * 64 + k] - mx);
                Sc[qt * 64 + k] = e;
                sum += e;
            }
            float inv = 1.f / sum;
            for (int k = 0; k < 49; ++k) Ps[qt * 64 + k] = f2bf(Sc[qt * 64 + k] * inv);
            for (int k = 49; k < 64; ++k) Ps[qt * 64 + k] = 0;
        }
    }
    __syncthreads();

    // out = P @ V : contraction 64 (2 ksteps), 32 output channels (2 ntiles)
    bf16x16 ap0 = load_frag_a(&Ps[(wv * 16) * 64 + 0], 64, lane);
    bf16x16 ap1 = load_frag_a(&Ps[(wv * 16) * 64 + 32], 64, lane);
#pragma unroll
    for (int nt = 0; nt < 2; ++nt) {
        bf16x16 bv0 = load_frag_b(&Vt[(nt * 16) * 64 + 0], 64, lane);
        bf16x16 bv1 = load_frag_b(&Vt[(nt * 16) * 64 + 32], 64, lane);
        f32x8 acc = {};
        acc = wmma_bf16(ap0, bv0, acc);
        acc = wmma_bf16(ap1, bv1, acc);
        AccF cf; cf.v = acc;
#pragma unroll
        for (int r = 0; r < 8; ++r) {
            int qt = wv * 16 + lh * 8 + r;
            if (qt < 49) {
                int bq = qt / 7, d = qt - bq * 7;
                int rr = a * 7 + bq, cc = c * 7 + d;
                if (shifted) { rr = (rr + 3) % 56; cc = (cc + 3) % 56; }
                size_t row = (size_t)b * 3136 + (size_t)rr * 56 + cc;
                attn[row * 128 + h * 32 + nt * 16 + ln] = f2bf(cf.f[r]);
            }
        }
    }
}

// ---------------------------------------------------------------------------
extern "C" void kernel_launch(void* const* d_in, const int* in_sizes, int n_in,
                              void* d_out, int out_size, void* d_ws, size_t ws_size,
                              hipStream_t stream) {
    const float* x      = (const float*)d_in[0];
    const float* ln1_g  = (const float*)d_in[1];
    const float* ln1_b  = (const float*)d_in[2];
    const float* w_qkv1 = (const float*)d_in[3];
    const float* b_qkv1 = (const float*)d_in[4];
    const float* w_o1   = (const float*)d_in[5];
    const float* b_o1   = (const float*)d_in[6];
    const float* tbl1   = (const float*)d_in[7];
    const float* ln2_g  = (const float*)d_in[8];
    const float* ln2_b  = (const float*)d_in[9];
    const float* w_mlp1 = (const float*)d_in[10];
    const float* b_mlp1 = (const float*)d_in[11];
    const float* ln3_g  = (const float*)d_in[12];
    const float* ln3_b  = (const float*)d_in[13];
    const float* w_qkv2 = (const float*)d_in[14];
    const float* b_qkv2 = (const float*)d_in[15];
    const float* w_o2   = (const float*)d_in[16];
    const float* b_o2   = (const float*)d_in[17];
    const float* tbl2   = (const float*)d_in[18];
    const float* ln4_g  = (const float*)d_in[19];
    const float* ln4_b  = (const float*)d_in[20];
    const float* w_mlp2 = (const float*)d_in[21];
    const float* b_mlp2 = (const float*)d_in[22];

    char* ws = (char*)d_ws;
    float*          x_cur  = (float*)ws;                              // 102,760,448 B
    unsigned short* ybf    = (unsigned short*)(ws + 102760448);       //  51,380,224 B
    unsigned short* qkvbf  = (unsigned short*)(ws + 154140672);       // 154,140,672 B
    unsigned short* attnbf = (unsigned short*)(ws + 308281344);       //  51,380,224 B

    dim3 gLN(25088), gG384(1568, 6), gG128(1568, 2), gAttn(16384);

    // block 1: W-MSA
    ln_bf16_kernel<<<gLN, 256, 0, stream>>>(x, ln1_g, ln1_b, ybf);
    gemm_bf16_kernel<true, false><<<gG384, 256, 0, stream>>>(ybf, w_qkv1, b_qkv1, nullptr, qkvbf, 384);
    win_attn_kernel<<<gAttn, 128, 0, stream>>>(qkvbf, tbl1, attnbf, 0);
    gemm_bf16_kernel<false, true><<<gG128, 256, 0, stream>>>(attnbf, w_o1, b_o1, x, x_cur, 128);
    // mlp1
    ln_bf16_kernel<<<gLN, 256, 0, stream>>>(x_cur, ln2_g, ln2_b, ybf);
    gemm_bf16_kernel<false, true><<<gG128, 256, 0, stream>>>(ybf, w_mlp1, b_mlp1, x_cur, x_cur, 128);
    // block 2: SW-MSA (roll folded into attention gather/scatter)
    ln_bf16_kernel<<<gLN, 256, 0, stream>>>(x_cur, ln3_g, ln3_b, ybf);
    gemm_bf16_kernel<true, false><<<gG384, 256, 0, stream>>>(ybf, w_qkv2, b_qkv2, nullptr, qkvbf, 384);
    win_attn_kernel<<<gAttn, 128, 0, stream>>>(qkvbf, tbl2, attnbf, 1);
    gemm_bf16_kernel<false, true><<<gG128, 256, 0, stream>>>(attnbf, w_o2, b_o2, x_cur, x_cur, 128);
    // mlp2 (final residual dropped, per reference)
    ln_bf16_kernel<<<gLN, 256, 0, stream>>>(x_cur, ln4_g, ln4_b, ybf);
    gemm_bf16_kernel<false, false><<<gG128, 256, 0, stream>>>(ybf, w_mlp2, b_mlp2, nullptr, d_out, 128);
}
